// MapLoss_6923487281605
// MI455X (gfx1250) — compile-verified, matrix operand order checked by reading, not run on hardware
//
#include <hip/hip_runtime.h>
#include <hip/hip_bf16.h>
#include <stdint.h>

// ---------------------------------------------------------------------------
// MapLoss forward for MI455X (gfx1250, wave32).
// Problem sizes fixed by the reference: B=256, C=8192, M=4 -> N_cross=1280.
// Memory-bound (~200MB total traffic ~ 9us @ 23.3TB/s); compute trivial.
// WMMA (v_wmma_f32_16x16x32_f16) is used as a 16x32 compare-count reduction
// engine with an all-ones B operand (layout-robust: only row->lane mapping
// of A and the C/D layout matter, both table-confirmed).
// ---------------------------------------------------------------------------

typedef __attribute__((ext_vector_type(16))) _Float16 v16h;
typedef __attribute__((ext_vector_type(8)))  _Float16 v8h;
typedef __attribute__((ext_vector_type(8)))  float    v8f;
typedef __attribute__((ext_vector_type(4)))  float    v4f;

#define B_      256
#define C_      8192
#define NROW    1280                 // B_ + 4*B_
#define HCROSS  5242880u             // (C_*NROW)/2
#define NI      2097152              // B_*C_
#define HI_     1048576u             // NI/2
#define PCAP    65536
#define MARGIN  0.02f

// workspace byte offsets (256B aligned)
#define OFF_SEQT   0ull
#define OFF_POST   41943040ull
#define OFF_ISEQ   52428800ull
#define OFF_PREC   60817408ull
#define OFF_PLIST  60850176ull
#define OFF_SORT   61112320ull
#define OFF_BUCKET 61374464ull
#define OFF_PCOUNT 61636864ull
#define WS_NEEDED  61637120ull

// ----------------------------- Threefry-2x32 (JAX-exact) -------------------
__host__ __device__ static inline uint32_t rotl32(uint32_t v, int r) {
  return (v << r) | (v >> (32 - r));
}

__host__ __device__ static inline void threefry2x32(uint32_t k0, uint32_t k1,
                                                    uint32_t x0, uint32_t x1,
                                                    uint32_t& o0, uint32_t& o1) {
  const uint32_t ks2 = k0 ^ k1 ^ 0x1BD11BDAu;
  x0 += k0; x1 += k1;
  x0 += x1; x1 = rotl32(x1, 13); x1 ^= x0;
  x0 += x1; x1 = rotl32(x1, 15); x1 ^= x0;
  x0 += x1; x1 = rotl32(x1, 26); x1 ^= x0;
  x0 += x1; x1 = rotl32(x1,  6); x1 ^= x0;
  x0 += k1; x1 += ks2 + 1u;
  x0 += x1; x1 = rotl32(x1, 17); x1 ^= x0;
  x0 += x1; x1 = rotl32(x1, 29); x1 ^= x0;
  x0 += x1; x1 = rotl32(x1, 16); x1 ^= x0;
  x0 += x1; x1 = rotl32(x1, 24); x1 ^= x0;
  x0 += ks2; x1 += k0 + 2u;
  x0 += x1; x1 = rotl32(x1, 13); x1 ^= x0;
  x0 += x1; x1 = rotl32(x1, 15); x1 ^= x0;
  x0 += x1; x1 = rotl32(x1, 26); x1 ^= x0;
  x0 += x1; x1 = rotl32(x1,  6); x1 ^= x0;
  x0 += k0; x1 += k1 + 3u;
  x0 += x1; x1 = rotl32(x1, 17); x1 ^= x0;
  x0 += x1; x1 = rotl32(x1, 29); x1 ^= x0;
  x0 += x1; x1 = rotl32(x1, 16); x1 ^= x0;
  x0 += x1; x1 = rotl32(x1, 24); x1 ^= x0;
  x0 += k1; x1 += ks2 + 4u;
  x0 += x1; x1 = rotl32(x1, 13); x1 ^= x0;
  x0 += x1; x1 = rotl32(x1, 15); x1 ^= x0;
  x0 += x1; x1 = rotl32(x1, 26); x1 ^= x0;
  x0 += x1; x1 = rotl32(x1,  6); x1 ^= x0;
  x0 += ks2; x1 += k0 + 5u;
  o0 = x0; o1 = x1;
}

// bits[f] of jax random_bits over a flat array of size 2*half (counts split in halves)
__device__ static inline uint32_t jax_bits(uint32_t k0, uint32_t k1, uint32_t f, uint32_t half) {
  uint32_t x0, x1, o0, o1;
  if (f < half) { x0 = f; x1 = f + half; } else { x0 = f - half; x1 = f; }
  threefry2x32(k0, k1, x0, x1, o0, o1);
  return (f < half) ? o0 : o1;
}

// jax.random.normal: uniform in [nextafter(-1,0), 1) -> sqrt(2)*erfinv(u)
__device__ static inline float jax_normal(uint32_t bits) {
  const float lo = -0.99999994f;                    // nextafter(-1,0) in f32
  float u01 = __uint_as_float((bits >> 9) | 0x3F800000u) - 1.0f;  // [0,1)
  float u = u01 * (1.0f - lo) + lo;
  u = fmaxf(lo, u);
  return 1.41421356237309515f * erfinvf(u);
}

// ----------------------------- K0: zero accumulators -----------------------
__global__ void MapK0(float* precArr, float* plist, float* sortedv, int* bucket, int* pcount) {
  int i = blockIdx.x * blockDim.x + threadIdx.x;
  int stride = gridDim.x * blockDim.x;
  for (int k = i; k < C_; k += stride) precArr[k] = 0.0f;
  for (int k = i; k < PCAP; k += stride) { plist[k] = 0.0f; sortedv[k] = 0.0f; }
  for (int k = i; k <= PCAP; k += stride) bucket[k] = 0;
  if (i == 0) *pcount = 0;
}

// ----------------------------- K1: threefry + margin + transpose -----------
// Reads long_out/long_tgt [N=1280, C=8192] coalesced; writes seqT/posT [C, N] coalesced.
__global__ void __launch_bounds__(256)
MapK1(const float* __restrict__ out0, const float* __restrict__ so,
      const float* __restrict__ tg0, const float* __restrict__ st,
      float* __restrict__ seqT, unsigned char* __restrict__ posT,
      uint32_t k0, uint32_t k1) {
  __shared__ float tile[32][33];
  __shared__ unsigned char ptile[32][33];
  const int tx = threadIdx.x & 31, ty = threadIdx.x >> 5;   // ty 0..7
  const int c0 = blockIdx.x * 32, n0 = blockIdx.y * 32;
#pragma unroll
  for (int r = 0; r < 4; ++r) {
    const int nl = ty + r * 8;
    const int n = n0 + nl, c = c0 + tx;
    float s, t;
    if (n < B_) { s = out0[(size_t)n * C_ + c]; t = tg0[(size_t)n * C_ + c]; }
    else        { s = so[(size_t)(n - B_) * C_ + c]; t = st[(size_t)(n - B_) * C_ + c]; }
    const uint32_t f = (uint32_t)c * NROW + (uint32_t)n;   // dev has shape [C, N]
    const float nrm = jax_normal(jax_bits(k0, k1, f, HCROSS));
    tile[nl][tx] = s - MARGIN * (fabsf(nrm) * (t - 0.5f));
    ptile[nl][tx] = (t > 0.5f) ? 1 : 0;
  }
  __syncthreads();
#pragma unroll
  for (int r = 0; r < 4; ++r) {
    const int cl = ty + r * 8;
    const int c = c0 + cl, n = n0 + tx;
    seqT[(size_t)c * NROW + n] = tile[tx][cl];
    posT[(size_t)c * NROW + n] = ptile[tx][cl];
  }
}

// ----------------------------- K2: per-class AP via WMMA counting ----------
// One block (4 waves) per class. Wave 0 computes, per positive p:
//   cglob = #{j: seq_j > v_p},  cpos = #{pos j: seq_j > v_p}
// via D = A x ones with v_wmma_f32_16x16x32_f16 (A = 16x32 0/1 indicators).
// Slot->element map K = h + ((h&8)?8:0) + kofs gives each lane two contiguous
// 8-element runs -> 4x ds_load_b128 (f32) + 2x ds_load_b128 (f16 mask) per
// chunk, and the indicator build is pure v_cmp + v_cndmask (branchless).
// AP contribution = (cpos+1)/(cglob+1); prec = sum / (P + 1e-5).
__global__ void __launch_bounds__(128)
MapK2(const float* __restrict__ seqT, const unsigned char* __restrict__ posT,
      float* __restrict__ precArr) {
  __shared__ __attribute__((aligned(16))) float    sseq[NROW];
  __shared__ __attribute__((aligned(16))) _Float16 sposh[NROW];  // 0.0h / 1.0h
  __shared__ float spv[128];
  __shared__ int sPsh;
  __shared__ float scnt[32];
  const int tid = threadIdx.x;
  const int c = blockIdx.x;
  const float* row = seqT + (size_t)c * NROW;
  const unsigned char* prow = posT + (size_t)c * NROW;
  for (int i = tid; i < NROW; i += 128) {
    __builtin_prefetch(row + i + 512, 0, 1);     // global_prefetch_b8
    sseq[i] = row[i];
    sposh[i] = prow[i] ? (_Float16)1.0f : (_Float16)0.0f;
  }
  __syncthreads();
  if (tid == 0) {                                // deterministic positive order
    int n = 0;
    for (int i = 0; i < NROW; ++i)
      if (sposh[i] != (_Float16)0.0f) { if (n < 128) spv[n] = sseq[i]; n++; }
    sPsh = n;
  }
  __syncthreads();
  const int rawP = sPsh;
  if (rawP == 0) return;                          // class absent -> precArr stays 0
  const int P = rawP > 128 ? 128 : rawP;

  if (tid < 32) {                                 // wave 0, EXEC all ones at WMMAs
    const int lane = tid;
    const int m = lane & 15;                      // A row (M) fed by lanes m and m+16
    const int kofs = (lane & 16) ? 8 : 0;
    const _Float16 one = (_Float16)1.0f;
    const _Float16 zero = (_Float16)0.0f;
    v16h bones;
#pragma unroll
    for (int h = 0; h < 16; ++h) bones[h] = one;  // B == ones (layout-free)
    float psum = 0.0f;
    const int ntiles = (P + 15) >> 4;
    for (int t = 0; t < ntiles; ++t) {
      const int p = t * 16 + m;
      const float vp = (p < P) ? spv[p] : __builtin_inff();   // padded rows count 0
      v8f cg = {}; v8f cp = {};
      for (int ch = 0; ch < NROW / 32; ++ch) {    // 40 chunks of K=32
        const int base = ch * 32 + kofs;
        // two contiguous runs of 8: [base, base+8) and [base+16, base+24)
        const v4f s0 = *(const v4f*)&sseq[base];
        const v4f s1 = *(const v4f*)&sseq[base + 4];
        const v4f s2 = *(const v4f*)&sseq[base + 16];
        const v4f s3 = *(const v4f*)&sseq[base + 20];
        const v8h q0 = *(const v8h*)&sposh[base];        // h = 0..7
        const v8h q1 = *(const v8h*)&sposh[base + 16];   // h = 8..15
        v16h aG, aP;
#pragma unroll
        for (int h = 0; h < 4; ++h) {
          const bool g0 = s0[h] > vp, g1 = s1[h] > vp;
          const bool g2 = s2[h] > vp, g3 = s3[h] > vp;
          aG[h]      = g0 ? one : zero;  aP[h]      = g0 ? q0[h]     : zero;
          aG[h + 4]  = g1 ? one : zero;  aP[h + 4]  = g1 ? q0[h + 4] : zero;
          aG[h + 8]  = g2 ? one : zero;  aP[h + 8]  = g2 ? q1[h]     : zero;
          aG[h + 12] = g3 ? one : zero;  aP[h + 12] = g3 ? q1[h + 4] : zero;
        }
        cg = __builtin_amdgcn_wmma_f32_16x16x32_f16(false, aG, false, bones,
                                                    (short)0, cg, false, false);
        cp = __builtin_amdgcn_wmma_f32_16x16x32_f16(false, aP, false, bones,
                                                    (short)0, cp, false, false);
      }
      // D layout (32-bit C/D 16x16): lane 0 holds M=0..7 @N=0, lane 16 holds M=8..15
      if (m == 0) {
        const int mofs = (lane & 16) ? 8 : 0;
#pragma unroll
        for (int vv = 0; vv < 8; ++vv) {
          scnt[mofs + vv]      = cg[vv];
          scnt[16 + mofs + vv] = cp[vv];
        }
      }
      __builtin_amdgcn_wave_barrier();            // order LDS write -> read in-wave
      if (p < P)
        psum += (scnt[16 + m] + 1.0f) / (scnt[m] + 1.0f);
      __builtin_amdgcn_wave_barrier();            // protect scnt before next tile
    }
#pragma unroll
    for (int off = 16; off > 0; off >>= 1)
      psum += __shfl_xor(psum, off, 32);
    if (lane == 0)
      precArr[c] = psum / ((float)rawP + 1e-5f);
  }
}

// ----------------------------- K3: interclass seq + positive list ----------
__global__ void MapK3(const float* __restrict__ out0, const float* __restrict__ tg0,
                      float* __restrict__ iseq, float* __restrict__ plist,
                      int* __restrict__ pcount, uint32_t k0, uint32_t k1) {
  const int i = blockIdx.x * blockDim.x + threadIdx.x;   // < NI
  const float s = out0[i], t = tg0[i];
  const float nrm = jax_normal(jax_bits(k0, k1, (uint32_t)i, HI_));
  const float v = s - MARGIN * (fabsf(nrm) * (t - 0.5f));
  iseq[i] = v;
  if (t > 0.5f) {
    int idx = atomicAdd(pcount, 1);
    if (idx < PCAP) plist[idx] = v;
  }
}

// ----------------------------- K4: sort positives by rank-counting ---------
__global__ void __launch_bounds__(256)
MapK4(const float* __restrict__ plist, const int* __restrict__ pcount,
      float* __restrict__ sortedv) {
  __shared__ float buf[256];
  int P = *pcount; if (P > PCAP) P = PCAP;
  const int i = blockIdx.x * blockDim.x + threadIdx.x;
  if (blockIdx.x * blockDim.x >= P) return;      // uniform per block
  const float v = (i < P) ? plist[i] : 0.0f;
  int cnt = 0;
  for (int base = 0; base < P; base += 256) {
    const int j = base + threadIdx.x;
    buf[threadIdx.x] = (j < P) ? plist[j] : -__builtin_inff();
    __syncthreads();
    const int lim = (P - base) < 256 ? (P - base) : 256;
    if (i < P)
      for (int t = 0; t < lim; ++t) cnt += (buf[t] > v) ? 1 : 0;
    __syncthreads();
  }
  if (i < P) sortedv[cnt] = v;                   // descending order by rank
}

// ----------------------------- K5: bucket histogram via binary search ------
__global__ void MapK5(const float* __restrict__ iseq, const float* __restrict__ sortedv,
                      const int* __restrict__ pcount, int* __restrict__ bucket) {
  const int i = blockIdx.x * blockDim.x + threadIdx.x;   // < NI
  int P = *pcount; if (P > PCAP) P = PCAP;
  const float e = iseq[i];
  int lo = 0, hi = P;                           // j = #{positives >= e}
  while (lo < hi) { int mid = (lo + hi) >> 1; if (sortedv[mid] >= e) lo = mid + 1; else hi = mid; }
  atomicAdd(&bucket[lo], 1);
}

// ----------------------------- K6: scan + deterministic final reduce -------
__global__ void __launch_bounds__(256)
MapK6(const float* __restrict__ precArr, const int* __restrict__ bucket,
      const int* __restrict__ pcount, float* __restrict__ out) {
  __shared__ float rf[256];
  __shared__ int   ri[256];
  __shared__ int   sbuf[256];
  const int tid = threadIdx.x;
  // cross-batch: fixed-order reduction of per-class precision
  float s = 0.0f; int np = 0;
  for (int c = tid; c < C_; c += 256) {
    const float p = precArr[c];
    if (p > 0.0f) { s += p; np++; }
  }
  rf[tid] = s; ri[tid] = np; __syncthreads();
  for (int off = 128; off > 0; off >>= 1) {
    if (tid < off) { rf[tid] += rf[tid + off]; ri[tid] += ri[tid + off]; }
    __syncthreads();
  }
  const float sumPrec = rf[0];
  const int nPres = ri[0];
  __syncthreads();
  // interclass: chunked inclusive scan of buckets; contribution (k+1)/(prefix+1)
  int P = *pcount; if (P > PCAP) P = PCAP;
  int carry = 0; float acc = 0.0f;
  for (int base = 0; base < P; base += 256) {
    const int k = base + tid;
    sbuf[tid] = (k < P) ? bucket[k] : 0;
    __syncthreads();
    for (int off = 1; off < 256; off <<= 1) {
      int v = sbuf[tid];
      if (tid >= off) v += sbuf[tid - off];
      __syncthreads();
      sbuf[tid] = v;
      __syncthreads();
    }
    const int incl = sbuf[tid] + carry;
    if (k < P) acc += (float)(k + 1) / (float)(incl + 1);
    carry += sbuf[255];
    __syncthreads();
  }
  rf[tid] = acc; __syncthreads();
  for (int off = 128; off > 0; off >>= 1) {
    if (tid < off) rf[tid] += rf[tid + off];
    __syncthreads();
  }
  if (tid == 0) {
    const float lossC = 1.0f - sumPrec / (float)(nPres > 0 ? nPres : 1);
    const float precI = rf[0] / ((float)P + 1e-5f);
    const float lossI = 1.0f - ((P > 0) ? precI : 0.0f);
    out[0] = 0.5f * lossC + 0.5f * lossI;       // (1-coef)*cross + coef*inter, coef=0.5
  }
}

// ----------------------------- launch --------------------------------------
extern "C" void kernel_launch(void* const* d_in, const int* in_sizes, int n_in,
                              void* d_out, int out_size, void* d_ws, size_t ws_size,
                              hipStream_t stream) {
  (void)in_sizes; (void)n_in; (void)out_size;
  if (ws_size < WS_NEEDED) return;   // workspace too small; cannot proceed

  const float* output        = (const float*)d_in[0];   // [256, 8192]
  const float* target        = (const float*)d_in[1];   // [256, 8192]
  const float* stored_output = (const float*)d_in[2];   // [1024, 8192]
  const float* stored_target = (const float*)d_in[3];   // [1024, 8192]
  float* outp = (float*)d_out;

  uint8_t* w = (uint8_t*)d_ws;
  float*         seqT    = (float*)(w + OFF_SEQT);
  unsigned char* posT    = (unsigned char*)(w + OFF_POST);
  float*         iseq    = (float*)(w + OFF_ISEQ);
  float*         precArr = (float*)(w + OFF_PREC);
  float*         plist   = (float*)(w + OFF_PLIST);
  float*         sortedv = (float*)(w + OFF_SORT);
  int*           bucket  = (int*)(w + OFF_BUCKET);
  int*           pcount  = (int*)(w + OFF_PCOUNT);

  // jax.random.split(jax.random.key(42)) -> (k_cross, k_inter), host-side Threefry
  uint32_t a0, b0, a1, b1;
  threefry2x32(0u, 42u, 0u, 2u, a0, b0);
  threefry2x32(0u, 42u, 1u, 3u, a1, b1);
  const uint32_t kc0 = a0, kc1 = a1;   // k_cross
  const uint32_t ki0 = b0, ki1 = b1;   // k_inter

  MapK0<<<256, 256, 0, stream>>>(precArr, plist, sortedv, bucket, pcount);
  MapK1<<<dim3(C_ / 32, NROW / 32), 256, 0, stream>>>(output, stored_output,
                                                      target, stored_target,
                                                      seqT, posT, kc0, kc1);
  MapK2<<<C_, 128, 0, stream>>>(seqT, posT, precArr);
  MapK3<<<NI / 256, 256, 0, stream>>>(output, target, iseq, plist, pcount, ki0, ki1);
  MapK4<<<PCAP / 256, 256, 0, stream>>>(plist, pcount, sortedv);
  MapK5<<<NI / 256, 256, 0, stream>>>(iseq, sortedv, pcount, bucket);
  MapK6<<<1, 256, 0, stream>>>(precArr, bucket, pcount, outp);
}
